// GNNModel_51582557225195
// MI455X (gfx1250) — matrix operand order repaired and checked by reading, not compiled, not run on hardware
//
#include <hip/hip_runtime.h>
#include <hip/hip_bf16.h>

typedef __attribute__((ext_vector_type(16))) __bf16 v16bf;
typedef __attribute__((ext_vector_type(8)))  float  v8f;

#define MB 4   // M-tiles (of 16 rows) per wave in the GEMM

// ---------- helpers ----------
__device__ __forceinline__ unsigned short f32_to_bf16_rne(float f) {
    unsigned u = __float_as_uint(f);
    unsigned rounding = 0x7FFFu + ((u >> 16) & 1u);
    u += rounding;
    return (unsigned short)(u >> 16);
}

// ---------- elementwise utility kernels ----------
__global__ void zero_f32_kernel(float* p, long n) {
    long i = (long)blockIdx.x * blockDim.x + threadIdx.x;
    if (i < n) p[i] = 0.0f;
}

__global__ void copy_f32_kernel(const float* __restrict__ src, float* __restrict__ dst, long n) {
    long i = (long)blockIdx.x * blockDim.x + threadIdx.x;
    if (i < n) dst[i] = src[i];
}

__global__ void f32_to_bf16_kernel(const float* __restrict__ in, unsigned short* __restrict__ out, long n) {
    long i = (long)blockIdx.x * blockDim.x + threadIdx.x;
    if (i < n) out[i] = f32_to_bf16_rne(in[i]);
}

// ---------- edge aggregation: agg[dst] += x[src] (float4 gather + atomic scatter) ----------
__global__ void edge_agg_kernel(const float* __restrict__ x,
                                const int* __restrict__ src,
                                const int* __restrict__ dst,
                                float* __restrict__ agg,
                                long total, int chunks, int K) {
    long t = (long)blockIdx.x * blockDim.x + threadIdx.x;
    if (t >= total) return;
    int e = (int)(t / chunks);
    int c = (int)(t % chunks) * 4;
    int s = src[e];
    int d = dst[e];
    const float4 v = *(const float4*)(x + (size_t)s * K + c);
    float* p = agg + (size_t)d * K + c;
    atomicAdd(p + 0, v.x);
    atomicAdd(p + 1, v.y);
    atomicAdd(p + 2, v.z);
    atomicAdd(p + 3, v.w);
}

// ---------- pack W (KxH fp32, row-major) into WMMA B-fragment order (bf16) ----------
// linear layout: (((nt*numKT + kt)*32) + lane)*16 + h
// B 32x16 layout: lanes 0-15 hold K=kt*32+0..7,16..23, lanes 16-31 hold K=kt*32+8..15,24..31; n = nt*16 + lane%16
__global__ void pack_w_kernel(const float* __restrict__ W, unsigned short* __restrict__ Wpk,
                              int K, int H) {
    long total = (long)K * H;
    long i = (long)blockIdx.x * blockDim.x + threadIdx.x;
    if (i >= total) return;
    int numKT = K >> 5;
    int h    = (int)(i & 15);
    int lane = (int)((i >> 4) & 31);
    long tile = i >> 9;
    int kt = (int)(tile % numKT);
    int nt = (int)(tile / numKT);
    int k = kt * 32 + ((h >> 3) << 4) + ((lane >= 16) ? 8 : 0) + (h & 7);
    int n = nt * 16 + (lane & 15);
    Wpk[i] = f32_to_bf16_rne(W[(size_t)k * H + n]);
}

// ---------- WMMA GEMM + fused bias + fused BN column stats ----------
// Y[N,H] = A_bf16[N,K] @ Wpk[K,H] + bias ; stats[c] += sum(Y[:,c]), stats[H+c] += sum(Y[:,c]^2)
// block = 256 threads = 8 waves; wave -> 4 vertically-stacked 16x16 tiles (64 rows x 16 cols),
// one shared B fragment per k-step. K is a template constant so every fragment load is
// base + compile-time immediate offset (lets the scheduler clause loads & pipeline WMMAs).
template <int K>
__global__ void gemm_wmma_stats_kernel(const unsigned short* __restrict__ A,
                                       const unsigned short* __restrict__ Wpk,
                                       const float* __restrict__ bias,
                                       float* __restrict__ Y,
                                       float* __restrict__ stats,
                                       int N, int H) {
    constexpr int numKT = K >> 5;
    int wave = threadIdx.x >> 5;
    int lane = threadIdx.x & 31;
    int mBase = blockIdx.x * (16 * MB);
    int nTile = blockIdx.y * 8 + wave;
    int nBase = nTile * 16;
    int kHalf = (lane >> 4) << 3;  // 0 or 8

    const unsigned short* bBase = Wpk + (((size_t)nTile * numKT) << 9) + ((size_t)lane << 4);

    v8f acc[MB];
#pragma unroll
    for (int mb = 0; mb < MB; ++mb) acc[mb] = (v8f){};

    union frag { uint4 u[2]; v16bf v; };

    if (mBase + 16 * MB <= N) {
        // fast path: single A base, all offsets are compile-time immediates
        const unsigned short* aBase = A + (size_t)(mBase + (lane & 15)) * K + kHalf;
#pragma unroll
        for (int kt = 0; kt < numKT; ++kt) {
            frag ub;
            ub.u[0] = *(const uint4*)(bBase + kt * 512);
            ub.u[1] = *(const uint4*)(bBase + kt * 512 + 8);
            frag ua[MB];
#pragma unroll
            for (int mb = 0; mb < MB; ++mb) {
                ua[mb].u[0] = *(const uint4*)(aBase + mb * (16 * K) + kt * 32);
                ua[mb].u[1] = *(const uint4*)(aBase + mb * (16 * K) + kt * 32 + 16);
            }
#pragma unroll
            for (int mb = 0; mb < MB; ++mb) {
                acc[mb] = __builtin_amdgcn_wmma_f32_16x16x32_bf16(
                    false, ua[mb].v, false, ub.v, (short)0, acc[mb], false, false);
            }
        }
    } else {
        // ragged tail: per-subtile clamped row pointers
        const unsigned short* aRow[MB];
#pragma unroll
        for (int mb = 0; mb < MB; ++mb) {
            int r = mBase + mb * 16 + (lane & 15);
            if (r > N - 1) r = N - 1;
            aRow[mb] = A + (size_t)r * K + kHalf;
        }
#pragma unroll
        for (int kt = 0; kt < numKT; ++kt) {
            frag ub;
            ub.u[0] = *(const uint4*)(bBase + kt * 512);
            ub.u[1] = *(const uint4*)(bBase + kt * 512 + 8);
#pragma unroll
            for (int mb = 0; mb < MB; ++mb) {
                frag ua;
                ua.u[0] = *(const uint4*)(aRow[mb] + kt * 32);
                ua.u[1] = *(const uint4*)(aRow[mb] + kt * 32 + 16);
                acc[mb] = __builtin_amdgcn_wmma_f32_16x16x32_bf16(
                    false, ua.v, false, ub.v, (short)0, acc[mb], false, false);
            }
        }
    }

    int col = nBase + (lane & 15);
    int rowOff = (lane >> 4) * 8;
    float bv = bias[col];
    float s = 0.0f, s2 = 0.0f;
#pragma unroll
    for (int mb = 0; mb < MB; ++mb) {
#pragma unroll
        for (int r = 0; r < 8; ++r) {
            int row = mBase + mb * 16 + r + rowOff;
            if (row < N) {
                float v = acc[mb][r] + bv;
                Y[(size_t)row * H + col] = v;
                s += v;
                s2 += v * v;
            }
        }
    }
    atomicAdd(&stats[col], s);
    atomicAdd(&stats[H + col], s2);
}

// runtime-K fallback (any K multiple of 32)
__global__ void gemm_wmma_stats_generic(const unsigned short* __restrict__ A,
                                        const unsigned short* __restrict__ Wpk,
                                        const float* __restrict__ bias,
                                        float* __restrict__ Y,
                                        float* __restrict__ stats,
                                        int N, int K, int H) {
    int wave = threadIdx.x >> 5;
    int lane = threadIdx.x & 31;
    int mBase = blockIdx.x * (16 * MB);
    int nTile = blockIdx.y * 8 + wave;
    int nBase = nTile * 16;
    int numKT = K >> 5;
    int kHalf = (lane >> 4) << 3;

    const unsigned short* aRow[MB];
#pragma unroll
    for (int mb = 0; mb < MB; ++mb) {
        int r = mBase + mb * 16 + (lane & 15);
        if (r > N - 1) r = N - 1;
        aRow[mb] = A + (size_t)r * K + kHalf;
    }
    const unsigned short* bPtr = Wpk + (((size_t)nTile * numKT) << 9) + ((size_t)lane << 4);

    v8f acc[MB];
#pragma unroll
    for (int mb = 0; mb < MB; ++mb) acc[mb] = (v8f){};

    for (int kt = 0; kt < numKT; ++kt) {
        union { uint4 u[2]; v16bf v; } ub;
        ub.u[0] = *(const uint4*)(bPtr);
        ub.u[1] = *(const uint4*)(bPtr + 8);
        bPtr += 512;
#pragma unroll
        for (int mb = 0; mb < MB; ++mb) {
            union { uint4 u[2]; v16bf v; } ua;
            ua.u[0] = *(const uint4*)(aRow[mb]);
            ua.u[1] = *(const uint4*)(aRow[mb] + 16);
            aRow[mb] += 32;
            acc[mb] = __builtin_amdgcn_wmma_f32_16x16x32_bf16(
                false, ua.v, false, ub.v, (short)0, acc[mb], false, false);
        }
    }

    int col = nBase + (lane & 15);
    int rowOff = (lane >> 4) * 8;
    float bv = bias[col];
    float s = 0.0f, s2 = 0.0f;
#pragma unroll
    for (int mb = 0; mb < MB; ++mb) {
#pragma unroll
        for (int r = 0; r < 8; ++r) {
            int row = mBase + mb * 16 + r + rowOff;
            if (row < N) {
                float v = acc[mb][r] + bv;
                Y[(size_t)row * H + col] = v;
                s += v;
                s2 += v * v;
            }
        }
    }
    atomicAdd(&stats[col], s);
    atomicAdd(&stats[H + col], s2);
}

// ---------- BN + ReLU (+ residual), optionally also re-initializing the aggregate buffer ----------
__global__ void bn_relu_res_kernel(const float* __restrict__ Y,
                                   const float* __restrict__ sums,
                                   const float* __restrict__ gamma,
                                   const float* __restrict__ beta,
                                   const float* __restrict__ prev,   // may be null
                                   float* __restrict__ Xout,
                                   float* __restrict__ aggInit,      // may be null (may alias Y)
                                   int N, int H) {
    long i = (long)blockIdx.x * blockDim.x + threadIdx.x;
    long total = (long)N * H;
    if (i >= total) return;
    int c = (int)(i % H);
    float invN = 1.0f / (float)N;
    float mean = sums[c] * invN;
    float var = sums[H + c] * invN - mean * mean;
    float v = (Y[i] - mean) * rsqrtf(var + 1e-5f) * gamma[c] + beta[c];
    v = fmaxf(v, 0.0f);
    if (prev) v += prev[i];
    Xout[i] = v;
    if (aggInit) aggInit[i] = v;   // elementwise in-place over Y is safe
}

// ---------- layer-3 variant: BN + ReLU + residual fused directly into global mean pool ----------
__global__ void bn_relu_pool_kernel(const float* __restrict__ Y,
                                    const float* __restrict__ sums,
                                    const float* __restrict__ gamma,
                                    const float* __restrict__ beta,
                                    const float* __restrict__ prev,
                                    const int* __restrict__ batch,
                                    float* __restrict__ pooled,
                                    float* __restrict__ cnt,
                                    int N, int H) {
    long i = (long)blockIdx.x * blockDim.x + threadIdx.x;
    long total = (long)N * H;
    if (i >= total) return;
    int c = (int)(i % H);
    int n = (int)(i / H);
    float invN = 1.0f / (float)N;
    float mean = sums[c] * invN;
    float var = sums[H + c] * invN - mean * mean;
    float v = (Y[i] - mean) * rsqrtf(var + 1e-5f) * gamma[c] + beta[c];
    v = fmaxf(v, 0.0f) + prev[i];
    int g = batch[n];
    atomicAdd(&pooled[(size_t)g * H + c], v);
    if (c == 0) atomicAdd(&cnt[g], 1.0f);
}

// ---------- MLP head: out[g] = relu(mean_g @ fcW1 + fcb1) @ fcW2 + fcb2 ----------
__global__ void head_kernel(const float* __restrict__ pooled,
                            const float* __restrict__ cnt,
                            const float* __restrict__ fcW1,
                            const float* __restrict__ fcb1,
                            const float* __restrict__ fcW2,
                            const float* __restrict__ fcb2,
                            float* __restrict__ out, int H) {
    __shared__ float pl[256];
    __shared__ float red[256];
    int g = blockIdx.x;
    int j = threadIdx.x;   // blockDim.x == H (256)
    float inv = 1.0f / fmaxf(cnt[g], 1.0f);
    pl[j] = pooled[(size_t)g * H + j] * inv;
    __syncthreads();
    float acc = fcb1[j];
    for (int c = 0; c < H; ++c) acc += pl[c] * fcW1[(size_t)c * H + j];
    acc = fmaxf(acc, 0.0f);
    red[j] = acc * fcW2[j];
    __syncthreads();
    for (int s = 128; s > 0; s >>= 1) {
        if (j < s) red[j] += red[j + s];
        __syncthreads();
    }
    if (j == 0) out[g] = red[0] + fcb2[0];
}

// ============================== host launcher ==============================
static inline size_t align256(size_t x) { return (x + 255) & ~(size_t)255; }
static inline int cdiv(long a, long b) { return (int)((a + b - 1) / b); }

extern "C" void kernel_launch(void* const* d_in, const int* in_sizes, int n_in,
                              void* d_out, int out_size, void* d_ws, size_t ws_size,
                              hipStream_t stream) {
    const int D = 128;
    const int N = in_sizes[0] / D;          // x: N x 128
    const int E = in_sizes[1] / 2;          // edge_index: 2 x E
    const int H = in_sizes[4] / D;          // W1: D x H  -> H = 256
    const int G = out_size;                 // output: G x 1

    const float* x     = (const float*)d_in[0];
    const int*   src   = (const int*)d_in[1];
    const int*   dst   = src + E;
    const int*   batch = (const int*)d_in[2];
    const float* W1 = (const float*)d_in[4];   const float* b1  = (const float*)d_in[5];
    const float* W2 = (const float*)d_in[6];   const float* b2  = (const float*)d_in[7];
    const float* W3 = (const float*)d_in[8];   const float* b3  = (const float*)d_in[9];
    const float* g1 = (const float*)d_in[10];  const float* be1 = (const float*)d_in[11];
    const float* g2 = (const float*)d_in[12];  const float* be2 = (const float*)d_in[13];
    const float* g3 = (const float*)d_in[14];  const float* be3 = (const float*)d_in[15];
    const float* fcW1 = (const float*)d_in[16]; const float* fcb1 = (const float*)d_in[17];
    const float* fcW2 = (const float*)d_in[18]; const float* fcb2 = (const float*)d_in[19];
    float* out = (float*)d_out;

    // ---- workspace carve (Ybuf = fp32 aggregate AND GEMM output; x3 never materialized) ----
    const size_t nh = (size_t)N * H;
    char* ws = (char*)d_ws;
    float* Ybuf = (float*)ws;                 ws += align256(nh * sizeof(float));
    float* X1   = (float*)ws;                 ws += align256(nh * sizeof(float));
    float* X2   = (float*)ws;                 ws += align256(nh * sizeof(float));
    unsigned short* Abf = (unsigned short*)ws; ws += align256(nh * sizeof(unsigned short));
    unsigned short* Wpk = (unsigned short*)ws; ws += align256((size_t)H * H * sizeof(unsigned short));
    float* stats  = (float*)ws;               ws += align256(2 * (size_t)H * sizeof(float));
    float* pooled = (float*)ws;               ws += align256((size_t)G * H * sizeof(float));
    float* cnt    = (float*)ws;               ws += align256((size_t)G * sizeof(float));
    (void)ws_size; (void)n_in;

    const int T = 256;

    // shared GEMM front-end for one layer: aggregate (already init'ed in Ybuf) -> bf16 -> GEMM+stats
    auto run_gemm = [&](const float* xin, int K, const float* W, const float* bias) {
        const long nk = (long)N * K;
        const int chunks = K >> 2;
        const long etotal = (long)E * chunks;
        edge_agg_kernel<<<cdiv(etotal, T), T, 0, stream>>>(xin, src, dst, Ybuf, etotal, chunks, K);
        f32_to_bf16_kernel<<<cdiv(nk, T), T, 0, stream>>>(Ybuf, Abf, nk);
        pack_w_kernel<<<cdiv((long)K * H, T), T, 0, stream>>>(W, Wpk, K, H);
        zero_f32_kernel<<<cdiv(2L * H, T), T, 0, stream>>>(stats, 2L * H);
        dim3 gg(cdiv(N, 16 * MB), H / 128);
        if (K == 128) {
            gemm_wmma_stats_kernel<128><<<gg, T, 0, stream>>>(Abf, Wpk, bias, Ybuf, stats, N, H);
        } else if (K == 256) {
            gemm_wmma_stats_kernel<256><<<gg, T, 0, stream>>>(Abf, Wpk, bias, Ybuf, stats, N, H);
        } else {
            gemm_wmma_stats_generic<<<gg, T, 0, stream>>>(Abf, Wpk, bias, Ybuf, stats, N, K, H);
        }
    };

    const long nhl = (long)nh;

    // ---- layer 1: agg init = x (explicit copy), output x1 (also re-seeds Ybuf as layer-2 agg) ----
    copy_f32_kernel<<<cdiv((long)N * D, T), T, 0, stream>>>(x, Ybuf, (long)N * D);
    run_gemm(x, D, W1, b1);
    bn_relu_res_kernel<<<cdiv(nhl, T), T, 0, stream>>>(Ybuf, stats, g1, be1,
                                                       nullptr, X1, Ybuf, N, H);

    // ---- layer 2: agg already seeded with x1; output x2 (re-seeds Ybuf as layer-3 agg) ----
    run_gemm(X1, H, W2, b2);
    bn_relu_res_kernel<<<cdiv(nhl, T), T, 0, stream>>>(Ybuf, stats, g2, be2,
                                                       X1, X2, Ybuf, N, H);

    // ---- layer 3: BN+ReLU+residual fused straight into the mean-pool accumulators ----
    run_gemm(X2, H, W3, b3);
    zero_f32_kernel<<<cdiv((long)G * H, T), T, 0, stream>>>(pooled, (long)G * H);
    zero_f32_kernel<<<1, T, 0, stream>>>(cnt, (long)G);
    bn_relu_pool_kernel<<<cdiv(nhl, T), T, 0, stream>>>(Ybuf, stats, g3, be3,
                                                        X2, batch, pooled, cnt, N, H);

    // ---- MLP head ----
    head_kernel<<<G, H, 0, stream>>>(pooled, cnt, fcW1, fcb1, fcW2, fcb2, out, H);
}
